// QuantConv_65884798321268
// MI455X (gfx1250) — compile-verified, matrix operand order checked by reading, not run on hardware
//
#include <hip/hip_runtime.h>
#include <stdint.h>

// ---- problem constants ----
#define B_   32
#define H_   56
#define W_   56
#define CIN  256
#define COUT 256
#define HW_  (H_*W_)
#define M_TOT (B_*HW_)

// ---- tiling ----
constexpr int BM  = 128;          // M rows per block
constexpr int BN  = 128;          // N cols per block (async kernel)
constexpr int FBN = 64;           // N cols per block (fallback kernel)
constexpr int KS  = 32;           // K per step
constexpr int LDR = 40;           // LDS row stride in bf16 (80 B, 16B-aligned, conflict-free)
constexpr int NSLICE = 9 * (CIN / KS);   // 72 k-steps

constexpr size_t A_ELEMS  = (size_t)B_ * H_ * W_ * CIN;        // 25,690,112
constexpr size_t A_BYTES  = A_ELEMS * 2;                       // one bf16 array
constexpr size_t BW_ROW   = (size_t)LDR * 2;                   // 80 B padded weight row
constexpr size_t BW_SLICE = (size_t)COUT * BW_ROW;             // 20,480 B per k-slice
constexpr size_t BW_BYTES = (size_t)NSLICE * BW_SLICE;         // 1,474,560
constexpr size_t WS_NEED  = 2 * A_BYTES + 2 * BW_BYTES;        // ~105.7 MB

typedef __attribute__((ext_vector_type(16))) __bf16 v16bf;
typedef __attribute__((ext_vector_type(8)))  float  v8f;

// ---------------- prep kernels: fp32 -> (hi, lo) bf16, run once per launch ----
__global__ __launch_bounds__(256)
void prep_input(const float* __restrict__ in, __bf16* __restrict__ hi,
                __bf16* __restrict__ lo) {
    size_t e = (size_t)blockIdx.x * 256 + threadIdx.x;
    float x  = in[e];
    __bf16 h = (__bf16)x;
    hi[e] = h;
    lo[e] = (__bf16)(x - (float)h);
}

// Weights into padded, transposed, LDS-ready layout: [slice][n][k(40)] bf16.
// src linear index e == (s*32+k)*256 + n  -> fully coalesced read.
__global__ __launch_bounds__(256)
void prep_weights(const float* __restrict__ kern, __bf16* __restrict__ hi,
                  __bf16* __restrict__ lo) {
    int e  = blockIdx.x * 256 + threadIdx.x;   // 0 .. 589823
    int n  = e & 255;
    int sk = e >> 8;
    int k  = sk & 31;
    int s  = sk >> 5;
    float w  = kern[e];
    __bf16 h = (__bf16)w;
    size_t d = ((size_t)s * COUT + n) * LDR + k;
    hi[d] = h;
    lo[d] = (__bf16)(w - (float)h);
}

// ---------------- helpers ----------------
__device__ __forceinline__ void async_b128(uint32_t lds, uint64_t gaddr) {
    // GLOBAL_LOAD_ASYNC_TO_LDS_B128: per-lane 16B global->LDS, tracked by ASYNCcnt.
    asm volatile("global_load_async_to_lds_b128 %0, %1, off"
                 :: "v"(lds), "v"(gaddr) : "memory");
}

// Load one 16x32 bf16 fragment (ISA A layout; B identical when stored [n][k]).
__device__ __forceinline__ v16bf ldfrag(const __bf16* lds, int row16, int lane) {
    const __bf16* p = lds + row16 * LDR + ((lane >> 4) << 3);
    v16bf f;
    ((uint4*)&f)[0] = *(const uint4*)(p);
    ((uint4*)&f)[1] = *(const uint4*)(p + 16);
    return f;
}

// ---------------- main kernel: async-staged, double-buffered, bf16x3 WMMA ----
__global__ __launch_bounds__(256)
void conv3x3_wmma_async(const __bf16* __restrict__ Ahg, const __bf16* __restrict__ Alg,
                        const __bf16* __restrict__ Bhg, const __bf16* __restrict__ Blg,
                        const float* __restrict__ bias, float* __restrict__ out)
{
    __shared__ __align__(16) __bf16 Ah[2][BM * LDR];
    __shared__ __align__(16) __bf16 Al[2][BM * LDR];
    __shared__ __align__(16) __bf16 Bh[2][BN * LDR];
    __shared__ __align__(16) __bf16 Bl[2][BN * LDR];

    const int t    = threadIdx.x;
    const int lane = t & 31;
    const int wave = t >> 5;
    const int wm   = wave >> 1;    // 0..3  (32-row M sub-block)
    const int wn   = wave & 1;     // 0..1  (64-col N sub-block)
    const int m0   = blockIdx.y * BM;
    const int n0   = blockIdx.x * BN;

    // ---- per-thread staging descriptors (constant across the k loop) ----
    // A: 4 issues; rows r=0..127, 16B chunks j=0..3, arr 0=hi 1=lo.
    int      aEl[4], aArr[4], aPix[4], aY[4], aX[4], aGo[4];
    #pragma unroll
    for (int i = 0; i < 4; ++i) {
        int lin = i * 256 + t;
        int arr = lin >> 9, c = lin & 511, r = c >> 2, j = c & 3;
        int m = m0 + r;
        int b = m / HW_, rem = m - b * HW_;
        int y = rem / W_, x = rem - y * W_;
        aEl[i] = r * LDR + j * 8;      // LDS element offset
        aArr[i] = arr;
        aPix[i] = m; aY[i] = y; aX[i] = x;
        aGo[i] = j * 16;               // byte offset within channel slice
    }
    // B: 5 issues; rows n=0..127, 16B chunks q=0..4 (80B padded row), arr hi/lo.
    int aBn[5], aBq[5], bArr[5];
    size_t bGo[5];
    #pragma unroll
    for (int i = 0; i < 5; ++i) {
        int lin = i * 256 + t;         // 0..1279
        int arr = lin >= 640;
        int c   = arr ? lin - 640 : lin;
        int n   = c / 5, q = c - n * 5;
        aBn[i] = n; aBq[i] = q; bArr[i] = arr;
        bGo[i] = (size_t)(n0 + n) * BW_ROW + (size_t)q * 16;
    }

    v8f acc[2][4] = {};

    auto stage = [&](int s, int buf) {
        const int f  = s >> 3, cs = s & 7;          // s = f*8 + cs
        const int dy = f / 3 - 1, dx = f % 3 - 1;
        // ---- weights: pure async copy (always in-bounds) ----
        #pragma unroll
        for (int i = 0; i < 5; ++i) {
            __bf16* lp = (bArr[i] ? &Bl[buf][0] : &Bh[buf][0]) + (aBn[i] * LDR + aBq[i] * 8);
            uint64_t g = (uint64_t)(uintptr_t)(bArr[i] ? Blg : Bhg)
                       + (size_t)s * BW_SLICE + bGo[i];
            async_b128((uint32_t)(uintptr_t)lp, g);
        }
        // ---- activations: async copy, or zero-fill for OOB taps ----
        const int dpix = dy * W_ + dx;
        const int c0b  = cs * KS * 2;               // channel-base byte offset
        #pragma unroll
        for (int i = 0; i < 4; ++i) {
            __bf16* lp = (aArr[i] ? &Al[buf][0] : &Ah[buf][0]) + aEl[i];
            const int iy = aY[i] + dy, ix = aX[i] + dx;
            if ((unsigned)iy < (unsigned)H_ && (unsigned)ix < (unsigned)W_) {
                uint64_t g = (uint64_t)(uintptr_t)(aArr[i] ? Alg : Ahg)
                           + (size_t)(aPix[i] + dpix) * (CIN * 2) + c0b + aGo[i];
                async_b128((uint32_t)(uintptr_t)lp, g);
            } else {
                *(uint4*)lp = make_uint4(0u, 0u, 0u, 0u);
            }
        }
    };

    stage(0, 0);

    for (int s = 0; s < NSLICE; ++s) {
        const int cur = s & 1;
        asm volatile("s_wait_asynccnt 0x0" ::: "memory");  // own async copies landed
        __syncthreads();                                   // everyone's copies visible
        if (s + 1 < NSLICE) stage(s + 1, cur ^ 1);

        const int rsel = lane & 15;
        v16bf ah[2], al[2], bh[4], bl[4];
        #pragma unroll
        for (int mi = 0; mi < 2; ++mi) {
            const int row = wm * 32 + mi * 16 + rsel;
            ah[mi] = ldfrag(Ah[cur], row, lane);
            al[mi] = ldfrag(Al[cur], row, lane);
        }
        #pragma unroll
        for (int ni = 0; ni < 4; ++ni) {
            const int col = wn * 64 + ni * 16 + rsel;
            bh[ni] = ldfrag(Bh[cur], col, lane);
            bl[ni] = ldfrag(Bl[cur], col, lane);
        }
        #pragma unroll
        for (int mi = 0; mi < 2; ++mi) {
            #pragma unroll
            for (int ni = 0; ni < 4; ++ni) {
                acc[mi][ni] = __builtin_amdgcn_wmma_f32_16x16x32_bf16(
                    false, ah[mi], false, bh[ni], (short)0, acc[mi][ni], false, false);
                acc[mi][ni] = __builtin_amdgcn_wmma_f32_16x16x32_bf16(
                    false, ah[mi], false, bl[ni], (short)0, acc[mi][ni], false, false);
                acc[mi][ni] = __builtin_amdgcn_wmma_f32_16x16x32_bf16(
                    false, al[mi], false, bh[ni], (short)0, acc[mi][ni], false, false);
            }
        }
    }

    // ---- epilogue: C/D layout — VGPR r: lanes 0-15 -> M=r, lanes 16-31 -> M=r+8
    const int ncol   = lane & 15;
    const int rowoff = (lane >> 4) * 8;
    #pragma unroll
    for (int mi = 0; mi < 2; ++mi) {
        #pragma unroll
        for (int ni = 0; ni < 4; ++ni) {
            const int n   = n0 + wn * 64 + ni * 16 + ncol;
            const float bv = bias[n];
            #pragma unroll
            for (int r = 0; r < 8; ++r) {
                const int m = m0 + wm * 32 + mi * 16 + rowoff + r;
                out[(size_t)m * COUT + n] = acc[mi][ni][r] + bv;
            }
        }
    }
}

// ---------------- fallback (self-contained, proven round-1 kernel) ----------
__global__ __launch_bounds__(256)
void conv3x3_wmma_fallback(const float* __restrict__ in,
                           const float* __restrict__ kern,
                           const float* __restrict__ bias,
                           float* __restrict__ out)
{
    __shared__ __align__(16) __bf16 Ah[BM * LDR];
    __shared__ __align__(16) __bf16 Al[BM * LDR];
    __shared__ __align__(16) __bf16 Bh[FBN * LDR];
    __shared__ __align__(16) __bf16 Bl[FBN * LDR];

    const int t = threadIdx.x, lane = t & 31, wave = t >> 5;
    const int wm = wave >> 1, wn = wave & 1;
    const int m0 = blockIdx.y * BM, n0 = blockIdx.x * FBN;

    v8f acc[2][2] = {};
    const int jA = t & 7, rA = t >> 3;
    const int nB = t & 63, kB0 = t >> 6;

    for (int f = 0; f < 9; ++f) {
        const int dy = f / 3 - 1, dx = f % 3 - 1;
        for (int c0 = 0; c0 < CIN; c0 += KS) {
            __syncthreads();
            #pragma unroll
            for (int p = 0; p < 8; ++p) {
                const int k = kB0 + p * 4;
                const float w = kern[((size_t)f * CIN + (c0 + k)) * COUT + (n0 + nB)];
                const __bf16 hi = (__bf16)w;
                Bh[nB * LDR + k] = hi;
                Bl[nB * LDR + k] = (__bf16)(w - (float)hi);
            }
            #pragma unroll
            for (int p = 0; p < 4; ++p) {
                const int r = rA + p * 32, m = m0 + r;
                const int b = m / HW_, rem = m - b * HW_;
                const int y = rem / W_, x = rem - y * W_;
                const int iy = y + dy, ix = x + dx;
                float4 v = make_float4(0.f, 0.f, 0.f, 0.f);
                if ((unsigned)iy < (unsigned)H_ && (unsigned)ix < (unsigned)W_)
                    v = *(const float4*)(in + (((size_t)b * H_ + iy) * W_ + ix) * CIN + c0 + jA * 4);
                __bf16* dh = &Ah[r * LDR + jA * 4];
                __bf16* dl = &Al[r * LDR + jA * 4];
                const __bf16 h0 = (__bf16)v.x, h1 = (__bf16)v.y,
                             h2 = (__bf16)v.z, h3 = (__bf16)v.w;
                dh[0] = h0; dh[1] = h1; dh[2] = h2; dh[3] = h3;
                dl[0] = (__bf16)(v.x - (float)h0); dl[1] = (__bf16)(v.y - (float)h1);
                dl[2] = (__bf16)(v.z - (float)h2); dl[3] = (__bf16)(v.w - (float)h3);
            }
            __syncthreads();
            const int rsel = lane & 15;
            v16bf ah[2], al[2], bh[2], bl[2];
            #pragma unroll
            for (int mi = 0; mi < 2; ++mi) {
                const int row = wm * 32 + mi * 16 + rsel;
                ah[mi] = ldfrag(Ah, row, lane);
                al[mi] = ldfrag(Al, row, lane);
            }
            #pragma unroll
            for (int ni = 0; ni < 2; ++ni) {
                const int col = wn * 32 + ni * 16 + rsel;
                bh[ni] = ldfrag(Bh, col, lane);
                bl[ni] = ldfrag(Bl, col, lane);
            }
            #pragma unroll
            for (int mi = 0; mi < 2; ++mi)
                #pragma unroll
                for (int ni = 0; ni < 2; ++ni) {
                    acc[mi][ni] = __builtin_amdgcn_wmma_f32_16x16x32_bf16(
                        false, ah[mi], false, bh[ni], (short)0, acc[mi][ni], false, false);
                    acc[mi][ni] = __builtin_amdgcn_wmma_f32_16x16x32_bf16(
                        false, ah[mi], false, bl[ni], (short)0, acc[mi][ni], false, false);
                    acc[mi][ni] = __builtin_amdgcn_wmma_f32_16x16x32_bf16(
                        false, al[mi], false, bh[ni], (short)0, acc[mi][ni], false, false);
                }
        }
    }
    const int ncol = lane & 15, rowoff = (lane >> 4) * 8;
    #pragma unroll
    for (int mi = 0; mi < 2; ++mi)
        #pragma unroll
        for (int ni = 0; ni < 2; ++ni) {
            const int n = n0 + wn * 32 + ni * 16 + ncol;
            const float bv = bias[n];
            #pragma unroll
            for (int r = 0; r < 8; ++r) {
                const int m = m0 + wm * 32 + mi * 16 + rowoff + r;
                out[(size_t)m * COUT + n] = acc[mi][ni][r] + bv;
            }
        }
}

// ---------------- host ----------------
extern "C" void kernel_launch(void* const* d_in, const int* in_sizes, int n_in,
                              void* d_out, int out_size, void* d_ws, size_t ws_size,
                              hipStream_t stream) {
    (void)in_sizes; (void)n_in; (void)out_size;
    const float* in   = (const float*)d_in[0];
    const float* kern = (const float*)d_in[1];
    const float* bias = (const float*)d_in[2];
    float* out        = (float*)d_out;

    if (ws_size >= WS_NEED && d_ws) {
        char*  ws  = (char*)d_ws;
        __bf16* Ahg = (__bf16*)(ws);
        __bf16* Alg = (__bf16*)(ws + A_BYTES);
        __bf16* Bhg = (__bf16*)(ws + 2 * A_BYTES);
        __bf16* Blg = (__bf16*)(ws + 2 * A_BYTES + BW_BYTES);
        prep_input  <<<(unsigned)(A_ELEMS / 256), 256, 0, stream>>>(in, Ahg, Alg);
        prep_weights<<<(9 * CIN * COUT) / 256, 256, 0, stream>>>(kern, Bhg, Blg);
        conv3x3_wmma_async<<<dim3(COUT / BN, M_TOT / BM), 256, 0, stream>>>(
            Ahg, Alg, Bhg, Blg, bias, out);
    } else {
        conv3x3_wmma_fallback<<<dim3(COUT / FBN, M_TOT / BM), 256, 0, stream>>>(
            in, kern, bias, out);
    }
}
